// NRDDecoder_76759655514755
// MI455X (gfx1250) — matrix operand order repaired and checked by reading, not compiled, not run on hardware
//
#include <hip/hip_runtime.h>
#include <hip/hip_bf16.h>
#include <stdint.h>

// ---------------------------------------------------------------------------
// NRDDecoder on gfx1250: all convs lowered to implicit GEMM on
// v_wmma_f32_16x16x32_bf16 (fp32 accumulate), NHWC bf16 activations.
// Dynamic-MLP hyper-params staged into LDS via TDM tensor_load_to_lds.
// ---------------------------------------------------------------------------

typedef __attribute__((ext_vector_type(16))) __bf16 bf16x16;
typedef __attribute__((ext_vector_type(8)))  float  f32x8;
typedef __attribute__((ext_vector_type(4)))  unsigned int u32x4;
typedef __attribute__((ext_vector_type(8)))  unsigned int u32x8;

union FragB16 { bf16x16 v; uint4 q[2]; };

__device__ __forceinline__ unsigned short f2bf(float f) {
  unsigned u = __float_as_uint(f);
  u += 0x7FFFu + ((u >> 16) & 1u);            // round-to-nearest-even
  return (unsigned short)(u >> 16);
}
__device__ __forceinline__ float bf2f(unsigned short h) {
  return __uint_as_float((unsigned)h << 16);
}

// ---------------------------------------------------------------------------
// Implicit-GEMM conv kernel. One wave computes a 32(M) x 32(N) output tile
// with 2x2 WMMA accumulators, modulo-rotated software pipeline (2 stages, no
// branches in steady state so each WMMA quad consumes loads issued one full
// iteration earlier -> partial s_wait_loadcnt, loads stay in flight).
// A = bf16 weights [Mpad, taps*C], K tap-major. In = bf16 NHWC [Bn,H,W,C].
// C is a power of two (Cshift = log2 C); taps==9 -> 3x3 SAME, taps==1 -> 1x1.
// #K-chunks (K/32) is even for every caller.
// Epilogue: val = acc*scale + shift (+bias), optional relu;
//           outB (bf16 NHWC, stride outStride) or outF (fp32 NHWC, stride Mreal).
// ---------------------------------------------------------------------------
__global__ __launch_bounds__(32)
void conv_gemm_wmma(const unsigned short* __restrict__ A,
                    const unsigned short* __restrict__ In,
                    int H, int W, int C, int Cshift, int taps, int Mreal,
                    const float* __restrict__ scale,
                    const float* __restrict__ shift,
                    const float* __restrict__ bias,
                    int relu,
                    unsigned short* __restrict__ outB, int outStride,
                    float* __restrict__ outF)
{
  const int lane  = threadIdx.x;           // 0..31
  const int nbase = blockIdx.x * 32;
  const int mbase = blockIdx.y * 32;
  const int K     = taps * C;
  const int Cmask = C - 1;
  const int HW    = H * W;

  // Per-lane column (N) coordinates + 3x3 validity mask for the B fragments.
  int ncol[2], py[2], px[2], vmask[2];
  size_t pixbase[2];
#pragma unroll
  for (int j = 0; j < 2; ++j) {
    int n = nbase + j * 16 + (lane & 15);
    ncol[j] = n;
    int b = n / HW;
    int r = n - b * HW;
    py[j] = r / W;
    px[j] = r - py[j] * W;
    pixbase[j] = (size_t)n * C;            // == ((b*H+py)*W+px)*C
    int m = 0;
    if (taps == 9) {
#pragma unroll
      for (int tap = 0; tap < 9; ++tap) {
        int yy = py[j] + tap / 3 - 1, xx = px[j] + tap % 3 - 1;
        if ((unsigned)yy < (unsigned)H && (unsigned)xx < (unsigned)W)
          m |= 1 << tap;
      }
    } else m = 1;
    vmask[j] = m;
  }
  const int kbA = (lane < 16) ? 0 : 8;     // 16-bit A frag K grouping (ISA 7.12.2)
  const int kbB = (lane < 16) ? 0 : 8;     // 16-bit B frag K grouping
  const int mA0 = mbase + (lane & 15);

  const unsigned short* pA[2];
  pA[0] = A + (size_t)mA0 * K + kbA;
  pA[1] = A + (size_t)(mA0 + 16) * K + kbA;

  f32x8 acc[2][2];
#pragma unroll
  for (int i = 0; i < 2; ++i)
#pragma unroll
    for (int j = 0; j < 2; ++j)
      acc[i][j] = (f32x8){0.f,0.f,0.f,0.f,0.f,0.f,0.f,0.f};

  auto load_frags = [&](FragB16 (&a)[2], FragB16 (&b)[2], int k0) {
    int tap = k0 >> Cshift;                // 0 for taps==1 (k0 < C)
    int c0  = k0 & Cmask;
    int toff = 0;
    if (taps == 9) {
      int dy = tap / 3 - 1, dx = tap % 3 - 1;
      toff = (dy * W + dx) * C;            // element offset, may be negative
    }
#pragma unroll
    for (int i = 0; i < 2; ++i) {
      const unsigned short* p = pA[i] + k0;
      a[i].q[0] = *(const uint4*)(p);
      a[i].q[1] = *(const uint4*)(p + 16);
    }
#pragma unroll
    for (int j = 0; j < 2; ++j) {
      if ((vmask[j] >> tap) & 1) {
        const unsigned short* p = In + (ptrdiff_t)pixbase[j] + toff + c0 + kbB;
        b[j].q[0] = *(const uint4*)(p);
        b[j].q[1] = *(const uint4*)(p + 16);
      } else {
        b[j].q[0] = make_uint4(0u,0u,0u,0u);
        b[j].q[1] = make_uint4(0u,0u,0u,0u);
      }
    }
  };
  auto wmma4 = [&](FragB16 (&a)[2], FragB16 (&b)[2]) {
#pragma unroll
    for (int i = 0; i < 2; ++i)
#pragma unroll
      for (int j = 0; j < 2; ++j)
        acc[i][j] = __builtin_amdgcn_wmma_f32_16x16x32_bf16(
            false, a[i].v, false, b[j].v, (short)0, acc[i][j], false, false);
  };

  // Modulo-rotated 2-stage pipeline. K/32 is even and K >= 64.
  FragB16 a0[2], b0[2], a1[2], b1[2];
  load_frags(a0, b0, 0);
  load_frags(a1, b1, 32);
  for (int k0 = 64; k0 < K; k0 += 64) {
    wmma4(a0, b0);
    load_frags(a0, b0, k0);
    wmma4(a1, b1);
    load_frags(a1, b1, k0 + 32);
  }
  wmma4(a0, b0);
  wmma4(a1, b1);

  // Epilogue: D layout -> lane has col N = ncol[j], 8 consecutive rows
  // starting at mrow0 = mbase + i*16 + (lane<16 ? 0 : 8).
#pragma unroll
  for (int i = 0; i < 2; ++i) {
    int mrow0 = mbase + i * 16 + ((lane < 16) ? 0 : 8);
    if (mrow0 + 8 > Mreal) continue;       // Mreal % 8 == 0 for all users
    float s[8], t[8];
#pragma unroll
    for (int r = 0; r < 8; ++r) {
      s[r] = scale ? scale[mrow0 + r] : 1.0f;
      t[r] = shift ? shift[mrow0 + r] : (bias ? bias[mrow0 + r] : 0.0f);
    }
#pragma unroll
    for (int j = 0; j < 2; ++j) {
      int n = ncol[j];
      float v[8];
#pragma unroll
      for (int r = 0; r < 8; ++r) {
        float val = acc[i][j][r] * s[r] + t[r];
        if (relu) val = fmaxf(val, 0.f);
        v[r] = val;
      }
      if (outB) {
        uint4 q;
        q.x = (unsigned)f2bf(v[0]) | ((unsigned)f2bf(v[1]) << 16);
        q.y = (unsigned)f2bf(v[2]) | ((unsigned)f2bf(v[3]) << 16);
        q.z = (unsigned)f2bf(v[4]) | ((unsigned)f2bf(v[5]) << 16);
        q.w = (unsigned)f2bf(v[6]) | ((unsigned)f2bf(v[7]) << 16);
        *(uint4*)(outB + (size_t)n * outStride + mrow0) = q;
      } else {
        float4 q0 = make_float4(v[0], v[1], v[2], v[3]);
        float4 q1 = make_float4(v[4], v[5], v[6], v[7]);
        float* op = outF + (size_t)n * Mreal + mrow0;
        *(float4*)(op)     = q0;
        *(float4*)(op + 4) = q1;
      }
    }
  }
}

// ---- OIHW fp32 3x3 weights -> bf16 [Mpad, 9*C], K = tap*C + c, zero padded M
__global__ __launch_bounds__(256)
void k_w3x3_to_bf16(const float* __restrict__ src, unsigned short* __restrict__ dst,
                    int Mreal, int Mpad, int C)
{
  size_t i = (size_t)blockIdx.x * 256 + threadIdx.x;
  size_t total = (size_t)Mpad * 9 * C;
  if (i >= total) return;
  int c = (int)(i % C);
  size_t t = i / C;
  int tap = (int)(t % 9);
  int m = (int)(t / 9);
  float v = 0.f;
  if (m < Mreal) v = src[((size_t)m * C + c) * 9 + tap];
  dst[i] = f2bf(v);                         // dst index == i by construction
}

// ---- flat fp32 -> bf16 (1x1 conv weights)
__global__ __launch_bounds__(256)
void k_f32_to_bf16(const float* __restrict__ src, unsigned short* __restrict__ dst,
                   size_t total)
{
  size_t i = (size_t)blockIdx.x * 256 + threadIdx.x;
  if (i >= total) return;
  dst[i] = f2bf(src[i]);
}

// ---- NCHW fp32 -> NHWC bf16
__global__ __launch_bounds__(256)
void k_nchw_to_nhwc_bf16(const float* __restrict__ src, unsigned short* __restrict__ dst,
                         int Bn, int C, int H, int W)
{
  size_t i = (size_t)blockIdx.x * 256 + threadIdx.x;
  size_t total = (size_t)Bn * C * H * W;
  if (i >= total) return;
  int c = (int)(i % C);
  size_t t = i / C;
  int x = (int)(t % W); t /= W;
  int y = (int)(t % H);
  int b = (int)(t / H);
  dst[i] = f2bf(src[(((size_t)b * C + c) * H + y) * W + x]);
}

// ---- global average pool of channels 0..511, broadcast into 512..1023
__global__ __launch_bounds__(256)
void k_pool_concat(unsigned short* __restrict__ xbuf /* [8][256 px][1024] */)
{
  int b = blockIdx.x;
  int t = threadIdx.x;
#pragma unroll
  for (int cc = 0; cc < 2; ++cc) {
    int c = t + cc * 256;                   // 0..511
    float s = 0.f;
    for (int p = 0; p < 256; ++p)
      s += bf2f(xbuf[(size_t)(b * 256 + p) * 1024 + c]);
    unsigned short g = f2bf(s * (1.f / 256.f));
    for (int p = 0; p < 256; ++p)
      xbuf[(size_t)(b * 256 + p) * 1024 + 512 + c] = g;
  }
}

// ---- 1x1 conv 48 -> 16, bf16 NHWC in, fp32 NHWC out (weights via LDS)
__global__ __launch_bounds__(256)
void k_c1b(const unsigned short* __restrict__ c1a, const float* __restrict__ w,
           const float* __restrict__ bias, float* __restrict__ out, int npix)
{
  __shared__ float sw[16 * 48];
  __shared__ float sb[16];
  int t = threadIdx.x;
  for (int i = t; i < 16 * 48; i += 256) sw[i] = w[i];
  if (t < 16) sb[t] = bias[t];
  __syncthreads();
  int p = blockIdx.x * 256 + t;
  if (p >= npix) return;
  float xin[48];
  const unsigned short* ip = c1a + (size_t)p * 48;
#pragma unroll
  for (int c = 0; c < 48; ++c) xin[c] = bf2f(ip[c]);
  float* op = out + (size_t)p * 16;
#pragma unroll
  for (int o = 0; o < 16; ++o) {
    float acc = sb[o];
#pragma unroll
    for (int c = 0; c < 48; ++c) acc += sw[o * 48 + c] * xin[c];
    op[o] = acc;
  }
}

// ---- per-pixel dynamic 3-layer MLP; one block per low-res pixel (8x8 hi-res).
// The 4928-float hyper-parameter blob is staged into LDS with the Tensor Data
// Mover (tensor_load_to_lds, TENSORcnt), issued once by wave 0.
__global__ __launch_bounds__(64)
void k_dynmlp(const float* __restrict__ params,   // [2048][4928]
              const float* __restrict__ c1,       // [131072][16] NHWC fp32
              const float* __restrict__ cat_scale,
              const float* __restrict__ cat_shift,
              float* __restrict__ out)            // [8][256][128][128]
{
  __shared__ float sp[4928];
  int blk = blockIdx.x;                     // == (b*16+Y)*16+X
  int b = blk >> 8, rem = blk & 255, Y = rem >> 4, X = rem & 15;
  const float* P = params + (size_t)blk * 4928;
  int t = threadIdx.x;

  if (t < 32) {                             // wave-uniform: only wave 0 issues TDM
    unsigned long long ga = (unsigned long long)(const void*)P;
    unsigned ldsoff = (unsigned)(uintptr_t)(void*)sp;  // flat LDS addr bits[31:0]
    // D# group 0: count=1 | lds_addr | global_addr[56:0] | type=2 (bits 127:126)
    u32x4 g0;
    g0.x = 1u;
    g0.y = ldsoff;
    g0.z = (unsigned)ga;
    g0.w = ((unsigned)(ga >> 32) & 0x01FFFFFFu) | (2u << 30);
    // D# group 1: data_size=4B; tensor_dim0=4928, tensor_dim1=1;
    //             tile_dim0=4928 (0x1340), tile_dim1=1; dim0_stride=4928.
    u32x8 g1;
    g1[0] = 0x00020000u;     // workgroup_mask=0, data_size=2 (4 bytes)
    g1[1] = 0x13400000u;     // tensor_dim0[15:0]=4928 in bits[31:16]
    g1[2] = 0x00010000u;     // tensor_dim0[31:16]=0, tensor_dim1[15:0]=1
    g1[3] = 0x13400000u;     // tensor_dim1[31:16]=0, tile_dim0=4928
    g1[4] = 0x00000001u;     // tile_dim1=1, tile_dim2=0
    g1[5] = 4928u;           // tensor_dim0_stride[31:0]
    g1[6] = 0u;              // stride[47:32]=0, tensor_dim1_stride[15:0]=0
    g1[7] = 0u;
    asm volatile("tensor_load_to_lds %0, %1" :: "s"(g0), "s"(g1) : "memory");
    __builtin_amdgcn_s_wait_tensorcnt(0);
  }
  __syncthreads();

  int py = t >> 3, px = t & 7;
  int y = Y * 8 + py, x = X * 8 + px;
  size_t pp = ((size_t)(b * 128 + y)) * 128 + x;

  float vin[18];
  vin[0] = (float)px * 0.125f;              // coord gx
  vin[1] = (float)py * 0.125f;              // coord gy
  const float* cp = c1 + pp * 16;
#pragma unroll
  for (int c = 0; c < 16; ++c) vin[2 + c] = cp[c];
#pragma unroll
  for (int c = 0; c < 18; ++c) vin[c] = vin[c] * cat_scale[c] + cat_shift[c];

  float h0[16];
#pragma unroll
  for (int o = 0; o < 16; ++o) {
    float a = sp[288 + o];
#pragma unroll
    for (int c = 0; c < 18; ++c) a += sp[o * 18 + c] * vin[c];
    h0[o] = fmaxf(a, 0.f);
  }
  float h1[16];
#pragma unroll
  for (int o = 0; o < 16; ++o) {
    float a = sp[560 + o];
#pragma unroll
    for (int c = 0; c < 16; ++c) a += sp[304 + o * 16 + c] * h0[c];
    h1[o] = fmaxf(a, 0.f);
  }
  for (int o = 0; o < 256; ++o) {
    float a = sp[4672 + o];
#pragma unroll
    for (int c = 0; c < 16; ++c) a += sp[576 + o * 16 + c] * h1[c];
    out[(((size_t)(b * 256 + o)) * 128 + y) * 128 + x] = a;
  }
}

// ---------------------------------------------------------------------------
extern "C" void kernel_launch(void* const* d_in, const int* in_sizes, int n_in,
                              void* d_out, int out_size, void* d_ws, size_t ws_size,
                              hipStream_t stream)
{
  (void)in_sizes; (void)n_in; (void)out_size; (void)ws_size;
  const float* res5         = (const float*)d_in[0];
  const float* res2         = (const float*)d_in[1];
  const float* bottleneck_w = (const float*)d_in[2];
  const float* bn1_scale    = (const float*)d_in[3];
  const float* bn1_shift    = (const float*)d_in[4];
  const float* cls0_w       = (const float*)d_in[5];
  const float* bn2_scale    = (const float*)d_in[6];
  const float* bn2_shift    = (const float*)d_in[7];
  const float* cls1_w       = (const float*)d_in[8];
  const float* cls1_b       = (const float*)d_in[9];
  const float* c1a_w        = (const float*)d_in[10];
  const float* bn3_scale    = (const float*)d_in[11];
  const float* bn3_shift    = (const float*)d_in[12];
  const float* c1b_w        = (const float*)d_in[13];
  const float* c1b_b        = (const float*)d_in[14];
  const float* cat_scale    = (const float*)d_in[15];
  const float* cat_shift    = (const float*)d_in[16];
  float* out = (float*)d_out;

  // Workspace layout (bytes, all offsets 256B aligned). Total ~118 MB.
  char* ws = (char*)d_ws;
  const size_t offA   = 0;                      // weight staging, 18,874,368 B max
  const size_t offX1  = offA   + 18874368;      // res5 NHWC bf16:   8,388,608
  const size_t offX1b = offX1  + 8388608;       // x1buf bf16 1024ch:4,194,304
  const size_t offX2b = offX1b + 4194304;       // x2buf bf16 1024ch:4,194,304
  const size_t offBIG = offX2b + 4194304;       // res2 NHWC bf16 (67,108,864) then params fp32 (40,370,176)
  const size_t offC1A = offBIG + 67108864;      // c1a bf16 [131072][48]: 12,582,912
  const size_t offC1  = offC1A + 12582912;      // c1 fp32 [131072][16]:  8,388,608

  unsigned short* A      = (unsigned short*)(ws + offA);
  unsigned short* X1     = (unsigned short*)(ws + offX1);
  unsigned short* X1BUF  = (unsigned short*)(ws + offX1b);
  unsigned short* X2BUF  = (unsigned short*)(ws + offX2b);
  unsigned short* R2     = (unsigned short*)(ws + offBIG);
  float*          PARAMS = (float*)(ws + offBIG);
  unsigned short* C1A    = (unsigned short*)(ws + offC1A);
  float*          C1     = (float*)(ws + offC1);

  // ---- Phase 1: low-level branch (uses BIG as res2 NHWC) -------------------
  {  // c1a weights: [48,256,3,3] -> bf16 [64, 2304]
    size_t total = (size_t)64 * 9 * 256;
    k_w3x3_to_bf16<<<dim3((unsigned)((total + 255) / 256)), 256, 0, stream>>>(
        c1a_w, A, 48, 64, 256);
  }
  {  // res2 NCHW -> NHWC bf16
    size_t total = (size_t)8 * 256 * 128 * 128;
    k_nchw_to_nhwc_bf16<<<dim3((unsigned)((total + 255) / 256)), 256, 0, stream>>>(
        res2, R2, 8, 256, 128, 128);
  }
  // GEMM4: c1a conv3x3 -> bn3 -> relu, out bf16 NHWC [131072][48]
  conv_gemm_wmma<<<dim3(131072 / 32, 64 / 32), 32, 0, stream>>>(
      A, R2, 128, 128, 256, 8, 9, 48, bn3_scale, bn3_shift, nullptr, 1,
      C1A, 48, nullptr);
  // c1b 1x1 48->16 (+bias), fp32 NHWC out
  k_c1b<<<dim3(131072 / 256), 256, 0, stream>>>(C1A, c1b_w, c1b_b, C1, 131072);

  // ---- Phase 2: hyper-network trunk ---------------------------------------
  {  // bottleneck weights: [512,2048,3,3] -> bf16 [512, 18432]
    size_t total = (size_t)512 * 9 * 2048;
    k_w3x3_to_bf16<<<dim3((unsigned)((total + 255) / 256)), 256, 0, stream>>>(
        bottleneck_w, A, 512, 512, 2048);
  }
  {  // res5 NCHW -> NHWC bf16
    size_t total = (size_t)8 * 2048 * 16 * 16;
    k_nchw_to_nhwc_bf16<<<dim3((unsigned)((total + 255) / 256)), 256, 0, stream>>>(
        res5, X1, 8, 2048, 16, 16);
  }
  // GEMM1: bottleneck conv3x3 -> bn1 -> relu, write channels 0..511 of x1buf
  conv_gemm_wmma<<<dim3(2048 / 32, 512 / 32), 32, 0, stream>>>(
      A, X1, 16, 16, 2048, 11, 9, 512, bn1_scale, bn1_shift, nullptr, 1,
      X1BUF, 1024, nullptr);
  // global pool -> broadcast into channels 512..1023
  k_pool_concat<<<dim3(8), 256, 0, stream>>>(X1BUF);

  {  // cls0 weights: [1024,1024,3,3] -> bf16 [1024, 9216]
    size_t total = (size_t)1024 * 9 * 1024;
    k_w3x3_to_bf16<<<dim3((unsigned)((total + 255) / 256)), 256, 0, stream>>>(
        cls0_w, A, 1024, 1024, 1024);
  }
  // GEMM2: cls0 conv3x3 -> bn2 -> relu, bf16 NHWC [2048][1024]
  conv_gemm_wmma<<<dim3(2048 / 32, 1024 / 32), 32, 0, stream>>>(
      A, X1BUF, 16, 16, 1024, 10, 9, 1024, bn2_scale, bn2_shift, nullptr, 1,
      X2BUF, 1024, nullptr);

  {  // cls1 weights: [4928,1024] -> bf16
    size_t total = (size_t)4928 * 1024;
    k_f32_to_bf16<<<dim3((unsigned)((total + 255) / 256)), 256, 0, stream>>>(
        cls1_w, A, total);
  }
  // GEMM3: cls1 1x1 + bias, fp32 NHWC params [2048][4928] (reuses BIG; R2 dead)
  conv_gemm_wmma<<<dim3(2048 / 32, 4928 / 32), 32, 0, stream>>>(
      A, X2BUF, 16, 16, 1024, 10, 1, 4928, nullptr, nullptr, cls1_b, 0,
      nullptr, 0, PARAMS);

  // ---- Phase 3: per-pixel dynamic MLP -> d_out ----------------------------
  k_dynmlp<<<dim3(2048), 64, 0, stream>>>(PARAMS, C1, cat_scale, cat_shift, out);
}
